// QuantLinear_27273042330117
// MI455X (gfx1250) — matrix operand order repaired and checked by reading, not compiled, not run on hardware
//
#include <hip/hip_runtime.h>
#include <hip/hip_bf16.h>

#define RANK    192
#define IN_DIM  4096
#define OUT_DIM 11008
#define M_TOT   8192        // B*S = 4*2048
#define QMAX_F  15.0f
#define EPS_F   1e-8f

typedef __bf16 bf16_t;
typedef __attribute__((ext_vector_type(16))) __bf16 v16bf;
typedef __attribute__((ext_vector_type(8)))  __bf16 v8bf;
typedef __attribute__((ext_vector_type(4)))  __bf16 v4bf;
typedef __attribute__((ext_vector_type(8)))  float  v8f;

// ---------------------------------------------------------------------------
// Kernel 1: convert activations f32 -> bf16 (grid-stride, float4 granular)
// ---------------------------------------------------------------------------
__global__ void cvt_x_bf16(const float* __restrict__ x,
                           bf16_t* __restrict__ xb, long n4)
{
    long i      = (long)blockIdx.x * blockDim.x + threadIdx.x;
    long stride = (long)gridDim.x * blockDim.x;
    const float4* xs = (const float4*)x;
    v4bf*         xd = (v4bf*)xb;
    for (; i < n4; i += stride) {
        float4 f = xs[i];
        v4bf o;
        o[0] = (bf16_t)f.x; o[1] = (bf16_t)f.y;
        o[2] = (bf16_t)f.z; o[3] = (bf16_t)f.w;
        xd[i] = o;
    }
}

// ---------------------------------------------------------------------------
// Kernel 2: one workgroup (256 thr) per output row o:
//   w_row = weight[o,:] + Bd[o,:RANK] @ Bu        (f32 VALU FMA)
//   per-row asymmetric 4-bit fake quant           (shfl+LDS min/max reduce)
//   store quantized-dequantized row as bf16
// ---------------------------------------------------------------------------
__global__ void __launch_bounds__(256, 2)
fuse_lora_quant(const float* __restrict__ weight,
                const float* __restrict__ Bd,
                const float* __restrict__ Bu,
                bf16_t* __restrict__ wq)
{
    const int o   = blockIdx.x;
    const int tid = threadIdx.x;
    const int c0  = tid * 16;                      // 256 thr * 16 cols = 4096

    __shared__ float bd[RANK];
    __shared__ float smn[8], smx[8];

    if (tid < RANK) bd[tid] = Bd[(size_t)o * RANK + tid];
    __syncthreads();

    // acc = weight row chunk
    float acc[16];
    const float4* wrow = (const float4*)(weight + (size_t)o * IN_DIM + c0);
    #pragma unroll
    for (int q = 0; q < 4; ++q) {
        float4 w = wrow[q];
        acc[q * 4 + 0] = w.x; acc[q * 4 + 1] = w.y;
        acc[q * 4 + 2] = w.z; acc[q * 4 + 3] = w.w;
    }

    // acc += Bd[o,r] * Bu[r, c0..c0+15]
    for (int r = 0; r < RANK; ++r) {
        float s = bd[r];
        const float4* bu = (const float4*)(Bu + (size_t)r * IN_DIM + c0);
        #pragma unroll
        for (int q = 0; q < 4; ++q) {
            float4 u = bu[q];
            acc[q * 4 + 0] = fmaf(s, u.x, acc[q * 4 + 0]);
            acc[q * 4 + 1] = fmaf(s, u.y, acc[q * 4 + 1]);
            acc[q * 4 + 2] = fmaf(s, u.z, acc[q * 4 + 2]);
            acc[q * 4 + 3] = fmaf(s, u.w, acc[q * 4 + 3]);
        }
    }

    // row min/max (clamped through 0)
    float mn = acc[0], mx = acc[0];
    #pragma unroll
    for (int j = 1; j < 16; ++j) { mn = fminf(mn, acc[j]); mx = fmaxf(mx, acc[j]); }
    #pragma unroll
    for (int off = 16; off > 0; off >>= 1) {       // wave32 butterfly
        mn = fminf(mn, __shfl_xor(mn, off, 32));
        mx = fmaxf(mx, __shfl_xor(mx, off, 32));
    }
    const int wid = tid >> 5, ln = tid & 31;
    if (ln == 0) { smn[wid] = mn; smx[wid] = mx; }
    __syncthreads();
    mn = smn[0]; mx = smx[0];
    #pragma unroll
    for (int w = 1; w < 8; ++w) { mn = fminf(mn, smn[w]); mx = fmaxf(mx, smx[w]); }
    mn = fminf(mn, 0.0f);
    mx = fmaxf(mx, 0.0f);

    const float scale = fmaxf((mx - mn) / QMAX_F, EPS_F);
    const float inv_s = 1.0f / scale;
    const float zero  = rintf(-mn * inv_s);

    // fake-quant + store bf16 (4x v4bf stores)
    v4bf* dst = (v4bf*)(wq + (size_t)o * IN_DIM + c0);
    #pragma unroll
    for (int q = 0; q < 4; ++q) {
        v4bf ov;
        #pragma unroll
        for (int j = 0; j < 4; ++j) {
            float v  = acc[q * 4 + j];
            float qv = fminf(fmaxf(rintf(v * inv_s) + zero, 0.0f), QMAX_F);
            ov[j] = (bf16_t)((qv - zero) * scale);
        }
        dst[q] = ov;
    }
}

// ---------------------------------------------------------------------------
// Kernel 3: bf16 WMMA GEMM  out[m,n] = sum_k x[m,k]*w[n,k] + bias[n]
//   block = 256 thr (8 waves), block tile 64(M) x 256(N)
//   wave  = 32x64 output (2x4 tiles of v_wmma_f32_16x16x32_bf16), K step 32
// ---------------------------------------------------------------------------
__global__ void __launch_bounds__(256, 1)
gemm_wmma_bf16(const bf16_t* __restrict__ X,   // [M_TOT, IN_DIM]
               const bf16_t* __restrict__ W,   // [OUT_DIM, IN_DIM] (quantized)
               const float*  __restrict__ bias,
               float* __restrict__ out)        // [M_TOT, OUT_DIM]
{
    const int lane  = threadIdx.x & 31;
    const int wave  = threadIdx.x >> 5;
    const int wm    = wave & 1;                 // 2 waves along M
    const int wn    = wave >> 1;                // 4 waves along N
    const int m0    = blockIdx.y * 64  + wm * 32;
    const int n0    = blockIdx.x * 256 + wn * 64;
    const int mlane = lane & 15;
    const int hi    = lane >> 4;
    const int kha   = hi * 16;                  // A-frag K-half offset (contiguous, matches B)
    const int khb   = hi * 16;                  // B-frag K-half offset (contiguous, matches A)

    v8f acc[2][4];
    #pragma unroll
    for (int mt = 0; mt < 2; ++mt)
        #pragma unroll
        for (int nt = 0; nt < 4; ++nt)
            #pragma unroll
            for (int e = 0; e < 8; ++e) acc[mt][nt][e] = 0.0f;

    const bf16_t* arow[2];
    arow[0] = X + (size_t)(m0 +      mlane) * IN_DIM;
    arow[1] = X + (size_t)(m0 + 16 + mlane) * IN_DIM;
    const bf16_t* brow[4];
    #pragma unroll
    for (int nt = 0; nt < 4; ++nt)
        brow[nt] = W + (size_t)(n0 + nt * 16 + mlane) * IN_DIM;

    for (int k0 = 0; k0 < IN_DIM; k0 += 32) {
        // A fragments: 16x32 bf16, per-lane two contiguous 8-elem (16B) chunks
        union { v16bf v; v8bf h[2]; } a[2];
        #pragma unroll
        for (int mt = 0; mt < 2; ++mt) {
            a[mt].h[0] = *(const v8bf*)(arow[mt] + k0 +      kha);
            a[mt].h[1] = *(const v8bf*)(arow[mt] + k0 +  8 + kha);
        }
        // B fragments: 32x16 bf16 (w rows are B columns), per-lane 32B chunk
        #pragma unroll
        for (int nt = 0; nt < 4; ++nt) {
            __builtin_prefetch(brow[nt] + k0 + khb + 128, 0, 3);
            v16bf b = *(const v16bf*)(brow[nt] + k0 + khb);
            #pragma unroll
            for (int mt = 0; mt < 2; ++mt)
                acc[mt][nt] = __builtin_amdgcn_wmma_f32_16x16x32_bf16(
                    /*neg_a=*/false, a[mt].v, /*neg_b=*/false, b,
                    /*c_mod=*/(short)0, acc[mt][nt],
                    /*reuse_a=*/false, /*reuse_b=*/false);
        }
    }

    // C/D layout: lanes 0-15 -> M = v, lanes 16-31 -> M = v+8 ; N = lane&15
    const int rbase = m0 + hi * 8;
    #pragma unroll
    for (int nt = 0; nt < 4; ++nt) {
        const int col = n0 + nt * 16 + mlane;
        const float bv = bias[col];
        #pragma unroll
        for (int mt = 0; mt < 2; ++mt) {
            #pragma unroll
            for (int v = 0; v < 8; ++v) {
                const int row = rbase + mt * 16 + v;
                out[(size_t)row * OUT_DIM + col] = acc[mt][nt][v] + bv;
            }
        }
    }
}

// ---------------------------------------------------------------------------
extern "C" void kernel_launch(void* const* d_in, const int* in_sizes, int n_in,
                              void* d_out, int out_size, void* d_ws, size_t ws_size,
                              hipStream_t stream)
{
    const float* x      = (const float*)d_in[0];   // [4,2048,4096]
    const float* weight = (const float*)d_in[1];   // [11008,4096]
    const float* Bd     = (const float*)d_in[2];   // [11008,192]
    const float* Bu     = (const float*)d_in[3];   // [192,4096]
    const float* bias   = (const float*)d_in[4];   // [11008]
    float* out          = (float*)d_out;           // [4,2048,11008]

    // workspace: x_bf16 (64 MB) | w_bf16 (86 MB)
    bf16_t* xb = (bf16_t*)d_ws;
    bf16_t* wb = xb + (size_t)M_TOT * IN_DIM;

    const long n4 = (long)M_TOT * IN_DIM / 4;
    cvt_x_bf16<<<2048, 256, 0, stream>>>(x, xb, n4);

    fuse_lora_quant<<<OUT_DIM, 256, 0, stream>>>(weight, Bd, Bu, wb);

    dim3 grid(OUT_DIM / 256, M_TOT / 64);          // (43, 128)
    gemm_wmma_bf16<<<grid, 256, 0, stream>>>(xb, wb, bias, out);
}
